// _UnimolPELayer_38070590112226
// MI455X (gfx1250) — matrix operand order repaired and checked by reading, not compiled, not run on hardware
//
#include <hip/hip_runtime.h>
#include <hip/hip_bf16.h>
#include <math.h>

// Problem dims
#define LQ  512
#define BQ  8
#define DQ  768
#define HQ  12
#define DHQ 64
#define PQ  128
#define BHQ 96          // B*H
#define MR  4096        // L*B

#define KSTEP 32        // K slab per stage (32 bf16 = 64B per row)
#define SLAB  (128 * KSTEP)   // elements per slab buffer

typedef __bf16 bf16_t;
typedef __attribute__((ext_vector_type(8)))  __bf16 v8bf;
typedef __attribute__((ext_vector_type(16))) __bf16 v16bf;
typedef __attribute__((ext_vector_type(8)))  float  v8f;

// ---------------------------------------------------------------------------
// WMMA fragment loaders (CDNA5 16x16x32 bf16 layouts, ISA 7.12.2)
// A (16x32, MxK): lane m = lane&15; half = lane>>4.
//   vec elems 0..7  = K[kb..kb+7],   elems 8..15 = K[kb+16..kb+23], kb = half*8
// B (32x16, KxN) from row-major [N,K]: lane n = lane&15;
//   vec elems 0..7  = K[kb..kb+7],   elems 8..15 = K[kb+16..kb+23], kb = half*8
// C/D (16x16 f32): elem r -> m = r + 8*half, n = lane&15
// ---------------------------------------------------------------------------
__device__ __forceinline__ v16bf load_frag_a(const bf16_t* base, int ld,
                                             int row, int k0, int lane) {
  const bf16_t* p = base + (size_t)row * ld + k0 + ((lane & 16) ? 8 : 0);
  v8bf lo = *(const v8bf*)p;
  v8bf hi = *(const v8bf*)(p + 16);
  return __builtin_shufflevector(lo, hi, 0,1,2,3,4,5,6,7,8,9,10,11,12,13,14,15);
}

__device__ __forceinline__ v16bf load_frag_b(const bf16_t* base, int ld,
                                             int row, int k0, int lane) {
  const bf16_t* p = base + (size_t)row * ld + k0 + ((lane & 16) ? 8 : 0);
  v8bf lo = *(const v8bf*)p;
  v8bf hi = *(const v8bf*)(p + 16);
  return __builtin_shufflevector(lo, hi, 0,1,2,3,4,5,6,7,8,9,10,11,12,13,14,15);
}

__device__ __forceinline__ v8f wmma_bf(v16bf a, v16bf b, v8f c) {
  return __builtin_amdgcn_wmma_f32_16x16x32_bf16(false, a, false, b,
                                                 (short)0, c, false, false);
}

// ---------------------------------------------------------------------------
// Async global->LDS staging (CDNA5 ASYNCcnt pipe, ISA 08_async_tensor).
// LDS byte address = low 32 bits of the generic pointer (aperture mapping).
// Async loads complete IN ORDER -> s_wait_asynccnt N gives double buffering.
// ---------------------------------------------------------------------------
__device__ __forceinline__ void async_ld_b128(bf16_t* lds_dst, const bf16_t* g) {
  unsigned lds = (unsigned)(uintptr_t)lds_dst;
  unsigned long long ga = (unsigned long long)(uintptr_t)g;
  asm volatile("global_load_async_to_lds_b128 %0, %1, off"
               :: "v"(lds), "v"(ga) : "memory");
}
__device__ __forceinline__ void wait_async0() {
  asm volatile("s_wait_asynccnt 0" ::: "memory");
}
__device__ __forceinline__ void wait_async4() {  // allow next slab in flight
  asm volatile("s_wait_asynccnt 4" ::: "memory");
}

// Stage a 128-row x 32-col bf16 slab (8KB) into LDS: 256 thr x 2 x 16B chunks.
__device__ __forceinline__ void stage_slab(const bf16_t* g, int ld, int baseRow,
                                           int k0, bf16_t* s) {
  int t = threadIdx.x;
#pragma unroll
  for (int c = 0; c < 2; ++c) {
    int chunk = t + c * 256;        // 0..511
    int row = chunk >> 2;           // 0..127
    int part = chunk & 3;           // 4 x 8 bf16 per row
    async_ld_b128(s + row * KSTEP + part * 8,
                  g + (size_t)(baseRow + row) * ld + k0 + part * 8);
  }
}

// ---------------------------------------------------------------------------
// Double-buffered LDS-staged GEMM core. Block = 256 thr = 8 waves, 4(M)x2(N).
// Block tile 128 x 128; wave tile 32 x 64 (2x4 WMMA accumulators).
// Per thread per slab-pair: 4 async b128 -> after issuing slab k+1,
// s_wait_asynccnt 4 guarantees slab k landed (in-order completion) while
// slab k+1 overlaps the 8 WMMAs.
// ---------------------------------------------------------------------------
__device__ __forceinline__ void gemm_core_staged(
    const bf16_t* A, int lda, const bf16_t* Bm, int ldb,
    int rowBlock, int colBlock, int K, int lane, int wave,
    bf16_t* sA, bf16_t* sB, v8f acc[2][4]) {
  int rloc = (wave & 3) * 32;
  int cloc = (wave >> 2) * 64;
  // prologue: slab 0 -> buffer 0
  stage_slab(A,  lda, rowBlock, 0, sA);
  stage_slab(Bm, ldb, colBlock, 0, sB);
  int buf = 0;
  for (int k0 = 0; k0 < K; k0 += KSTEP) {
    int nbuf = buf ^ 1;
    if (k0 + KSTEP < K) {
      stage_slab(A,  lda, rowBlock, k0 + KSTEP, sA + nbuf * SLAB);
      stage_slab(Bm, ldb, colBlock, k0 + KSTEP, sB + nbuf * SLAB);
      if (k0 + 2 * KSTEP < K) {     // warm L2 for the slab after next
        __builtin_prefetch(A  + (size_t)(rowBlock + (threadIdx.x >> 1)) * lda + k0 + 2 * KSTEP, 0, 1);
        __builtin_prefetch(Bm + (size_t)(colBlock + (threadIdx.x >> 1)) * ldb + k0 + 2 * KSTEP, 0, 1);
      }
      wait_async4();                // slab k landed; slab k+1 still in flight
    } else {
      wait_async0();
    }
    __syncthreads();                // all waves' slab k visible in LDS
    const bf16_t* cA = sA + buf * SLAB;
    const bf16_t* cB = sB + buf * SLAB;
    v16bf a0 = load_frag_a(cA, KSTEP, rloc      + (lane & 15), 0, lane);
    v16bf a1 = load_frag_a(cA, KSTEP, rloc + 16 + (lane & 15), 0, lane);
    v16bf b0 = load_frag_b(cB, KSTEP, cloc      + (lane & 15), 0, lane);
    v16bf b1 = load_frag_b(cB, KSTEP, cloc + 16 + (lane & 15), 0, lane);
    v16bf b2 = load_frag_b(cB, KSTEP, cloc + 32 + (lane & 15), 0, lane);
    v16bf b3 = load_frag_b(cB, KSTEP, cloc + 48 + (lane & 15), 0, lane);
    acc[0][0] = wmma_bf(a0, b0, acc[0][0]);
    acc[1][0] = wmma_bf(a1, b0, acc[1][0]);
    acc[0][1] = wmma_bf(a0, b1, acc[0][1]);
    acc[1][1] = wmma_bf(a1, b1, acc[1][1]);
    acc[0][2] = wmma_bf(a0, b2, acc[0][2]);
    acc[1][2] = wmma_bf(a1, b2, acc[1][2]);
    acc[0][3] = wmma_bf(a0, b3, acc[0][3]);
    acc[1][3] = wmma_bf(a1, b3, acc[1][3]);
    __syncthreads();                // slab k consumed before it is re-staged
    buf = nbuf;
  }
}

// Direct-load 32x32 core (kept for attn@V where N=64)
__device__ __forceinline__ void gemm_core_32x32(const bf16_t* A, int lda,
                                                const bf16_t* Bm, int ldb,
                                                int rowBase, int colBase, int K,
                                                int lane, v8f acc2[2][2]) {
  for (int k0 = 0; k0 < K; k0 += 32) {
    if (k0 + 32 < K) {
      __builtin_prefetch(A  + (size_t)(rowBase + (lane & 15)) * lda + k0 + 32, 0, 1);
      __builtin_prefetch(Bm + (size_t)(colBase + (lane & 15)) * ldb + k0 + 32, 0, 1);
    }
    v16bf a0 = load_frag_a(A,  lda, rowBase      + (lane & 15), k0, lane);
    v16bf a1 = load_frag_a(A,  lda, rowBase + 16 + (lane & 15), k0, lane);
    v16bf b0 = load_frag_b(Bm, ldb, colBase      + (lane & 15), k0, lane);
    v16bf b1 = load_frag_b(Bm, ldb, colBase + 16 + (lane & 15), k0, lane);
    acc2[0][0] = wmma_bf(a0, b0, acc2[0][0]);
    acc2[0][1] = wmma_bf(a0, b1, acc2[0][1]);
    acc2[1][0] = wmma_bf(a1, b0, acc2[1][0]);
    acc2[1][1] = wmma_bf(a1, b1, acc2[1][1]);
  }
}

#define ACC_INIT24 v8f z = {0.f,0.f,0.f,0.f,0.f,0.f,0.f,0.f}; \
                   v8f acc[2][4] = {{z,z,z,z},{z,z,z,z}};
#define ACC_INIT22 v8f z = {0.f,0.f,0.f,0.f,0.f,0.f,0.f,0.f}; \
                   v8f acc2[2][2] = {{z,z},{z,z}};

#define SMEM_DECL __shared__ __align__(16) bf16_t sA[2 * SLAB]; \
                  __shared__ __align__(16) bf16_t sB[2 * SLAB];

// ---------------------------------------------------------------------------
// Fused GEMM: C = A[M,K] @ B[N,K]^T + bias, per-mode epilogue.
// mode 0: QKV      -> scatter q/k [B,H,L,DH] bf16, v -> vT [B,H,DH,L] bf16
// mode 1: PE proj  -> pe_q / pe_k [H,P,DH] bf16
// mode 2: out proj -> x1 = resid + C          (f32)
// mode 3: FFN1     -> gelu(C)                 (bf16)
// mode 4: FFN2     -> x_out = resid + C       (f32)
// ---------------------------------------------------------------------------
__global__ __launch_bounds__(256) void gemm_fused_kernel(
    const bf16_t* __restrict__ A, const bf16_t* __restrict__ Bm,
    const float* __restrict__ bias, const float* __restrict__ resid,
    float* __restrict__ outF, bf16_t* __restrict__ outB,
    bf16_t* __restrict__ outX, bf16_t* __restrict__ outY,
    bf16_t* __restrict__ outZ, int M, int N, int K, int mode) {
  SMEM_DECL;
  int lane = threadIdx.x & 31;
  int wave = threadIdx.x >> 5;
  int rowBlock = blockIdx.x * 128;
  int colBlock = blockIdx.y * 128;
  ACC_INIT24;
  gemm_core_staged(A, K, Bm, K, rowBlock, colBlock, K, lane, wave, sA, sB, acc);

  int rloc = (wave & 3) * 32, cloc = (wave >> 2) * 64;
#pragma unroll
  for (int tm = 0; tm < 2; ++tm)
#pragma unroll
    for (int tn = 0; tn < 4; ++tn)
#pragma unroll
      for (int r = 0; r < 8; ++r) {
        int m = rowBlock + rloc + tm * 16 + ((lane >> 4) << 3) + r;
        int n = colBlock + cloc + tn * 16 + (lane & 15);
        float v = acc[tm][tn][r] + bias[n];
        if (mode == 0) {
          int which = n / DQ, rem = n % DQ;
          int h = rem / DHQ, dh = rem % DHQ;
          int l = m >> 3, b = m & 7;
          if (which == 2) {
            outZ[(((size_t)(b * HQ + h)) * DHQ + dh) * LQ + l] = (bf16_t)v;
          } else {
            bf16_t* dst = (which == 0) ? outX : outY;
            dst[(((size_t)(b * HQ + h)) * LQ + l) * DHQ + dh] = (bf16_t)v;
          }
        } else if (mode == 1) {
          int which = n / DQ, rem = n % DQ;
          int h = rem / DHQ, dh = rem % DHQ;
          bf16_t* dst = (which == 0) ? outX : outY;
          dst[((size_t)h * PQ + m) * DHQ + dh] = (bf16_t)v;
        } else if (mode == 2) {
          outF[(size_t)m * N + n] = v + resid[(size_t)m * N + n];
        } else if (mode == 3) {
          float g = 0.5f * v * (1.f + erff(v * 0.70710678118654752f));
          outB[(size_t)m * N + n] = (bf16_t)g;
        } else {
          outF[(size_t)m * N + n] = v + resid[(size_t)m * N + n];
        }
      }
}

// kp[bh] = k[bh] (L x DH) @ pe_q[h] (P x DH)^T -> [L,P] f32  (also qp)
__global__ __launch_bounds__(256) void pe_gemm_kernel(
    const bf16_t* __restrict__ Xb, const bf16_t* __restrict__ PEb,
    float* __restrict__ Out) {
  SMEM_DECL;
  int lane = threadIdx.x & 31, wave = threadIdx.x >> 5;
  int bh = blockIdx.z, h = bh % HQ;
  const bf16_t* A  = Xb  + (size_t)bh * LQ * DHQ;
  const bf16_t* Bm = PEb + (size_t)h * PQ * DHQ;
  int rowBlock = blockIdx.x * 128;
  ACC_INIT24;
  gemm_core_staged(A, DHQ, Bm, DHQ, rowBlock, 0, DHQ, lane, wave, sA, sB, acc);
  float* O = Out + (size_t)bh * LQ * PQ;
  int rloc = (wave & 3) * 32, cloc = (wave >> 2) * 64;
#pragma unroll
  for (int tm = 0; tm < 2; ++tm)
#pragma unroll
    for (int tn = 0; tn < 4; ++tn)
#pragma unroll
      for (int r = 0; r < 8; ++r) {
        int m = rowBlock + rloc + tm * 16 + ((lane >> 4) << 3) + r;
        int n = cloc + tn * 16 + (lane & 15);
        O[(size_t)m * PQ + n] = acc[tm][tn][r];
      }
}

// scores[bh,i,j] = (q.k + kp[j,idx] + qp[i,idx]) * inv + edge  == edge_out
__global__ __launch_bounds__(256) void scores_kernel(
    const bf16_t* __restrict__ Qb, const bf16_t* __restrict__ Kb,
    const float* __restrict__ KP, const float* __restrict__ QP,
    const int* __restrict__ bdist, const float* __restrict__ edge,
    float* __restrict__ S) {
  SMEM_DECL;
  int lane = threadIdx.x & 31, wave = threadIdx.x >> 5;
  int bh = blockIdx.z, b = bh / HQ;
  const bf16_t* A  = Qb + (size_t)bh * LQ * DHQ;
  const bf16_t* Bm = Kb + (size_t)bh * LQ * DHQ;
  int rowBlock = blockIdx.x * 128;
  int colBlock = blockIdx.y * 128;
  ACC_INIT24;
  gemm_core_staged(A, DHQ, Bm, DHQ, rowBlock, colBlock, DHQ, lane, wave, sA, sB, acc);
  const float* kp = KP + (size_t)bh * LQ * PQ;
  const float* qp = QP + (size_t)bh * LQ * PQ;
  const float inv = 0.125f;  // 1/sqrt(DH)
  int rloc = (wave & 3) * 32, cloc = (wave >> 2) * 64;
#pragma unroll
  for (int tm = 0; tm < 2; ++tm)
#pragma unroll
    for (int tn = 0; tn < 4; ++tn)
#pragma unroll
      for (int r = 0; r < 8; ++r) {
        int i = rowBlock + rloc + tm * 16 + ((lane >> 4) << 3) + r;
        int j = colBlock + cloc + tn * 16 + (lane & 15);
        int idx = bdist[((size_t)i * LQ + j) * BQ + b];
        float t = acc[tm][tn][r] + kp[(size_t)j * PQ + idx]
                                 + qp[(size_t)i * PQ + idx];
        size_t e = (size_t)bh * LQ * LQ + (size_t)i * LQ + j;
        S[e] = t * inv + edge[e];
      }
}

// o[bh] = attn[bh] (L x L) @ vT[bh] (DH x L)^T -> [L,DH] -> scatter [L*B, D]
__global__ __launch_bounds__(256) void attnv_kernel(
    const bf16_t* __restrict__ At, const bf16_t* __restrict__ VT,
    bf16_t* __restrict__ O) {
  int lane = threadIdx.x & 31, wave = threadIdx.x >> 5;
  int bh = blockIdx.z, b = bh / HQ, h = bh % HQ;
  const bf16_t* A  = At + (size_t)bh * LQ * LQ;
  const bf16_t* Bm = VT + (size_t)bh * DHQ * LQ;
  int rowBase = blockIdx.x * 128 + (wave & 3) * 32;
  int colBase = (wave >> 2) * 32;  // N = 64
  ACC_INIT22;
  gemm_core_32x32(A, LQ, Bm, LQ, rowBase, colBase, LQ, lane, acc2);
#pragma unroll
  for (int tm = 0; tm < 2; ++tm)
#pragma unroll
    for (int tn = 0; tn < 2; ++tn)
#pragma unroll
      for (int r = 0; r < 8; ++r) {
        int l  = rowBase + tm * 16 + ((lane >> 4) << 3) + r;
        int dh = colBase + tn * 16 + (lane & 15);
        O[((size_t)l * BQ + b) * DQ + h * DHQ + dh] = (bf16_t)acc2[tm][tn][r];
      }
}

// ----------------------------- reductions ---------------------------------
__device__ __forceinline__ float wave_sum(float v) {
  for (int o = 16; o; o >>= 1) v += __shfl_xor(v, o, 32);
  return v;
}
__device__ __forceinline__ float wave_max(float v) {
  for (int o = 16; o; o >>= 1) v = fmaxf(v, __shfl_xor(v, o, 32));
  return v;
}
__device__ __forceinline__ float block_sum(float v, float* sh) {
  v = wave_sum(v);
  __syncthreads();
  if ((threadIdx.x & 31) == 0) sh[threadIdx.x >> 5] = v;
  __syncthreads();
  float r = sh[0];
#pragma unroll
  for (int i = 1; i < 8; ++i) r += sh[i];
  return r;
}
__device__ __forceinline__ float block_max(float v, float* sh) {
  v = wave_max(v);
  __syncthreads();
  if ((threadIdx.x & 31) == 0) sh[threadIdx.x >> 5] = v;
  __syncthreads();
  float r = sh[0];
#pragma unroll
  for (int i = 1; i < 8; ++i) r = fmaxf(r, sh[i]);
  return r;
}

// softmax over rows of S (len 512), emit bf16 probs
__global__ __launch_bounds__(256) void softmax_kernel(
    const float* __restrict__ S, bf16_t* __restrict__ A) {
  __shared__ float sh[8];
  size_t row = blockIdx.x;
  const float* s = S + row * LQ;
  int t = threadIdx.x;
  float v0 = s[t], v1 = s[t + 256];
  float mx = block_max(fmaxf(v0, v1), sh);
  float e0 = __expf(v0 - mx), e1 = __expf(v1 - mx);
  float rs = 1.f / block_sum(e0 + e1, sh);
  bf16_t* a = A + row * LQ;
  a[t]       = (bf16_t)(e0 * rs);
  a[t + 256] = (bf16_t)(e1 * rs);
}

// LayerNorm over D=768, emit bf16
__global__ __launch_bounds__(256) void ln_kernel(
    const float* __restrict__ X, const float* __restrict__ w,
    const float* __restrict__ bb, bf16_t* __restrict__ Y) {
  __shared__ float sh[8];
  size_t row = blockIdx.x;
  const float* x = X + row * DQ;
  int t = threadIdx.x;
  float v0 = x[t], v1 = x[t + 256], v2 = x[t + 512];
  float mu = block_sum(v0 + v1 + v2, sh) * (1.f / DQ);
  float d0 = v0 - mu, d1 = v1 - mu, d2 = v2 - mu;
  float var = block_sum(d0 * d0 + d1 * d1 + d2 * d2, sh) * (1.f / DQ);
  float r = rsqrtf(var + 1e-5f);
  bf16_t* y = Y + row * DQ;
  y[t]       = (bf16_t)(d0 * r * w[t]       + bb[t]);
  y[t + 256] = (bf16_t)(d1 * r * w[t + 256] + bb[t + 256]);
  y[t + 512] = (bf16_t)(d2 * r * w[t + 512] + bb[t + 512]);
}

__global__ void cvt_kernel(const float* __restrict__ src,
                           bf16_t* __restrict__ dst, size_t n) {
  size_t i = (size_t)blockIdx.x * blockDim.x + threadIdx.x;
  size_t stride = (size_t)gridDim.x * blockDim.x;
  for (; i < n; i += stride) dst[i] = (bf16_t)src[i];
}

// ---------------------------------------------------------------------------
extern "C" void kernel_launch(void* const* d_in, const int* in_sizes, int n_in,
                              void* d_out, int out_size, void* d_ws,
                              size_t ws_size, hipStream_t stream) {
  (void)in_sizes; (void)n_in; (void)out_size; (void)ws_size;
  const float* x      = (const float*)d_in[0];
  const float* edge   = (const float*)d_in[1];
  const int*   bdist  = (const int*)d_in[2];
  const float* W_in   = (const float*)d_in[3];
  const float* b_in   = (const float*)d_in[4];
  const float* pe_emb = (const float*)d_in[5];
  const float* W_out  = (const float*)d_in[6];
  const float* b_out  = (const float*)d_in[7];
  const float* W1     = (const float*)d_in[8];
  const float* b1     = (const float*)d_in[9];
  const float* W2     = (const float*)d_in[10];
  const float* b2     = (const float*)d_in[11];
  const float* ln1w   = (const float*)d_in[12];
  const float* ln1b   = (const float*)d_in[13];
  const float* ln2w   = (const float*)d_in[14];
  const float* ln2b   = (const float*)d_in[15];

  float* x_out    = (float*)d_out;
  float* edge_out = x_out + (size_t)LQ * BQ * DQ;  // == pre-softmax scores

  // workspace layout (256B aligned)
  char* base = (char*)d_ws;
  size_t off = 0;
  auto take = [&](size_t bytes) {
    size_t o = off;
    off += (bytes + 255) & ~(size_t)255;
    return base + o;
  };
  bf16_t* xn_bf   = (bf16_t*)take((size_t)MR * DQ * 2);
  bf16_t* win_bf  = (bf16_t*)take((size_t)3 * DQ * DQ * 2);
  bf16_t* wout_bf = (bf16_t*)take((size_t)DQ * DQ * 2);
  bf16_t* w1_bf   = (bf16_t*)take((size_t)4 * DQ * DQ * 2);
  bf16_t* w2_bf   = (bf16_t*)take((size_t)4 * DQ * DQ * 2);
  bf16_t* pe_bf   = (bf16_t*)take((size_t)PQ * DQ * 2);
  bf16_t* q_bf    = (bf16_t*)take((size_t)BHQ * LQ * DHQ * 2);
  bf16_t* k_bf    = (bf16_t*)take((size_t)BHQ * LQ * DHQ * 2);
  bf16_t* vt_bf   = (bf16_t*)take((size_t)BHQ * DHQ * LQ * 2);
  bf16_t* peq_bf  = (bf16_t*)take((size_t)HQ * PQ * DHQ * 2);
  bf16_t* pek_bf  = (bf16_t*)take((size_t)HQ * PQ * DHQ * 2);
  float*  kp_f    = (float*) take((size_t)BHQ * LQ * PQ * 4);
  float*  qp_f    = (float*) take((size_t)BHQ * LQ * PQ * 4);
  bf16_t* attn_bf = (bf16_t*)take((size_t)BHQ * LQ * LQ * 2);
  bf16_t* o_bf    = (bf16_t*)take((size_t)MR * DQ * 2);
  float*  x1_f    = (float*) take((size_t)MR * DQ * 4);
  bf16_t* h_bf    = (bf16_t*)take((size_t)MR * DQ * 2);
  bf16_t* g_bf    = (bf16_t*)take((size_t)MR * 4 * DQ * 2);

  // 1) weight/PE conversions to bf16
  cvt_kernel<<<256, 256, 0, stream>>>(W_in,   win_bf,  (size_t)3 * DQ * DQ);
  cvt_kernel<<<256, 256, 0, stream>>>(W_out,  wout_bf, (size_t)DQ * DQ);
  cvt_kernel<<<256, 256, 0, stream>>>(W1,     w1_bf,   (size_t)4 * DQ * DQ);
  cvt_kernel<<<256, 256, 0, stream>>>(W2,     w2_bf,   (size_t)4 * DQ * DQ);
  cvt_kernel<<<256, 256, 0, stream>>>(pe_emb, pe_bf,   (size_t)PQ * DQ);

  // 2) LN1
  ln_kernel<<<MR, 256, 0, stream>>>(x, ln1w, ln1b, xn_bf);

  // 3) QKV projection (M=4096, N=2304, K=768), scatter q/k/vT
  gemm_fused_kernel<<<dim3(32, 18), 256, 0, stream>>>(
      xn_bf, win_bf, b_in, nullptr, nullptr, nullptr,
      q_bf, k_bf, vt_bf, MR, 3 * DQ, DQ, 0);

  // 4) PE projection (M=128, N=1536, K=768) -> pe_q / pe_k [H,P,DH]
  gemm_fused_kernel<<<dim3(1, 12), 256, 0, stream>>>(
      pe_bf, win_bf, b_in, nullptr, nullptr, nullptr,
      peq_bf, pek_bf, nullptr, PQ, 2 * DQ, DQ, 1);

  // 5) kp = k @ pe_q^T, qp = q @ pe_k^T  (per head: 512x128, K=64)
  pe_gemm_kernel<<<dim3(4, 1, BHQ), 256, 0, stream>>>(k_bf, peq_bf, kp_f);
  pe_gemm_kernel<<<dim3(4, 1, BHQ), 256, 0, stream>>>(q_bf, pek_bf, qp_f);

  // 6) scores = (q.k^T + gathered PE)*inv + edge -> d_out edge region
  scores_kernel<<<dim3(4, 4, BHQ), 256, 0, stream>>>(
      q_bf, k_bf, kp_f, qp_f, bdist, edge, edge_out);

  // 7) softmax rows -> bf16 attn
  softmax_kernel<<<BHQ * LQ, 256, 0, stream>>>(edge_out, attn_bf);

  // 8) o = attn @ v  (per head: 512x64, K=512), scatter to [L*B, D]
  attnv_kernel<<<dim3(4, 1, BHQ), 256, 0, stream>>>(attn_bf, vt_bf, o_bf);

  // 9) x1 = x + o @ W_out^T + b_out  (M=4096, N=768, K=768)
  gemm_fused_kernel<<<dim3(32, 6), 256, 0, stream>>>(
      o_bf, wout_bf, b_out, x, x1_f, nullptr,
      nullptr, nullptr, nullptr, MR, DQ, DQ, 2);

  // 10) LN2
  ln_kernel<<<MR, 256, 0, stream>>>(x1_f, ln2w, ln2b, h_bf);

  // 11) g = gelu(h @ W1^T + b1)  (M=4096, N=3072, K=768)
  gemm_fused_kernel<<<dim3(32, 24), 256, 0, stream>>>(
      h_bf, w1_bf, b1, nullptr, nullptr, g_bf,
      nullptr, nullptr, nullptr, MR, 4 * DQ, DQ, 3);

  // 12) x_out = x1 + g @ W2^T + b2  (M=4096, N=768, K=3072) -> d_out
  gemm_fused_kernel<<<dim3(32, 6), 256, 0, stream>>>(
      g_bf, w2_bf, b2, x1_f, x_out, nullptr,
      nullptr, nullptr, nullptr, MR, DQ, 4 * DQ, 4);
}
